// FasterRCNN_919123001705
// MI455X (gfx1250) — matrix operand order repaired and checked by reading, not compiled, not run on hardware
//
#include <hip/hip_runtime.h>
#include <hip/hip_bf16.h>
#include <math.h>

typedef __attribute__((ext_vector_type(16))) _Float16 v16h;
typedef __attribute__((ext_vector_type(8)))  float    v8f;
typedef __attribute__((ext_vector_type(4)))  unsigned int u32x4;
typedef __attribute__((ext_vector_type(8)))  int      i32x8;
typedef __attribute__((ext_vector_type(4)))  int      i32x4;

#define CIN   256
#define HID   256
#define HH    14
#define WW    14
#define NPIX  196          // 14*14
#define NBOX  1764         // 9*196
#define MT_CONV 13         // ceil(196/16)
#define MT_HEAD 111        // ceil(1764/16)
#define KC_CONV 72         // 2304/32
#define KC_HEAD 8          // 256/32
#define GSTRIDE 208        // padded pixel stride (13*16) -> unconditional stores
#define CONV_B_BYTES (KC_CONV * 32 * 16 * 2)   // 73728 B per N-tile
#define CONV_B_DW    (CONV_B_BYTES / 4)        // 18432 dwords

#if defined(__has_builtin)
#if __has_builtin(__builtin_amdgcn_tensor_load_to_lds)
#define HAVE_TDM 1
#else
#define HAVE_TDM 0
#endif
#else
#define HAVE_TDM 0
#endif

// ---------------------------------------------------------------------------
// Pack kernels: write f16 operand fragments in the exact WMMA VGPR layout
// (ISA 7.12.2). One thread per 32-byte lane-slot.
// A (16-bit, 16x32): lane half 0 -> K 0..7,16..23 ; half 1 -> K 8..15,24..31
// B (16-bit, 32x16): lane half 0 -> K 0..7,16..23 ; half 1 -> K 8..15,24..31
// ---------------------------------------------------------------------------
__global__ __launch_bounds__(256)
void pack_convA(const float* __restrict__ features, _Float16* __restrict__ packA) {
  const int slot = blockIdx.x * blockDim.x + threadIdx.x;
  if (slot >= MT_CONV * KC_CONV * 32) return;
  const int lane  = slot & 31;
  const int kc    = (slot >> 5) % KC_CONV;
  const int mtile = slot / (KC_CONV * 32);
  const int half = lane >> 4, l15 = lane & 15;
  const int p  = mtile * 16 + l15;
  const int py = p / WW, px = p % WW;
  const int tap = kc >> 3, ky = tap / 3, kx = tap % 3;
  const int cin0 = (kc & 7) * 32;
  const int fy = py + ky - 1, fx = px + kx - 1;
  const bool inb = (p < NPIX) && (fy >= 0) && (fy < HH) && (fx >= 0) && (fx < WW);
  const int fbase = inb ? (fy * WW + fx) : 0;
  v16h a{};
#pragma unroll
  for (int j = 0; j < 8; ++j) {
    const int ca = cin0 + half * 8 + j;
    a[j]     = (_Float16)(inb ? features[ca * NPIX + fbase] : 0.0f);
    a[j + 8] = (_Float16)(inb ? features[(ca + 16) * NPIX + fbase] : 0.0f);
  }
  *(v16h*)(packA + (size_t)slot * 16) = a;
}

__global__ __launch_bounds__(256)
void pack_convB(const float* __restrict__ rpn_w, _Float16* __restrict__ packB) {
  const int slot = blockIdx.x * blockDim.x + threadIdx.x;
  if (slot >= 16 * KC_CONV * 32) return;
  const int lane  = slot & 31;
  const int kc    = (slot >> 5) % KC_CONV;
  const int ntile = slot / (KC_CONV * 32);
  const int half = lane >> 4, l15 = lane & 15;
  const int oc = ntile * 16 + l15;
  const int tap = kc >> 3;
  const int cin0 = (kc & 7) * 32;
  v16h b{};
#pragma unroll
  for (int i = 0; i < 16; ++i)
    b[i] = (_Float16)rpn_w[oc * (CIN * 9) + (cin0 + half * 8 + (i & 7) + (i >> 3) * 16) * 9 + tap];
  *(v16h*)(packB + (size_t)slot * 16) = b;
}

__global__ __launch_bounds__(256)
void pack_h1B(const float* __restrict__ h1_w, _Float16* __restrict__ packB) {
  const int slot = blockIdx.x * blockDim.x + threadIdx.x;
  if (slot >= 16 * KC_HEAD * 32) return;
  const int lane  = slot & 31;
  const int kc    = (slot >> 5) % KC_HEAD;
  const int ntile = slot / (KC_HEAD * 32);
  const int half = lane >> 4, l15 = lane & 15;
  const int col = ntile * 16 + l15;
  v16h b{};
#pragma unroll
  for (int i = 0; i < 16; ++i)
    b[i] = (_Float16)h1_w[col * HID + kc * 32 + half * 8 + (i & 7) + (i >> 3) * 16];
  *(v16h*)(packB + (size_t)slot * 16) = b;
}

__global__ __launch_bounds__(256)
void pack_headA(const float* __restrict__ rois, _Float16* __restrict__ packA) {
  const int slot = blockIdx.x * blockDim.x + threadIdx.x;
  if (slot >= MT_HEAD * KC_HEAD * 32) return;
  const int lane  = slot & 31;
  const int kc    = (slot >> 5) % KC_HEAD;
  const int mtile = slot / (KC_HEAD * 32);
  const int half = lane >> 4, l15 = lane & 15;
  const int row = mtile * 16 + l15;
  const bool rv = row < NBOX;
  const int rbase = (rv ? row : 0) * HID;
  v16h a{};
#pragma unroll
  for (int j = 0; j < 8; ++j) {
    const int k = kc * 32 + half * 8 + j;
    a[j]     = (_Float16)(rv ? rois[rbase + k] : 0.0f);
    a[j + 8] = (_Float16)(rv ? rois[rbase + k + 16] : 0.0f);
  }
  *(v16h*)(packA + (size_t)slot * 16) = a;
}

// ---------------------------------------------------------------------------
// TDM: 1-D tile load (CONV_B_DW dwords) from global to LDS offset 0.
// D# per ISA 8.3/8.4: group0 {count=1 | lds_addr | global_addr | type=2},
// group1 {data_size=4B | tensor_dim0 | tensor_dim1=1 | tile_dim0 | stride0}.
// ---------------------------------------------------------------------------
__device__ __forceinline__ void tdm_load_to_lds0(const void* src) {
#if HAVE_TDM
  const unsigned nd = CONV_B_DW;            // 18432, fits 16-bit tile_dim0
  const unsigned long long ga = (unsigned long long)(uintptr_t)src;
  u32x4 g0;
  g0[0] = 1u;                                // count=1, user descriptor
  g0[1] = 0u;                                // lds_addr = 0 (sole __shared__)
  g0[2] = (unsigned)ga;                      // global_addr[31:0]
  g0[3] = (unsigned)((ga >> 32) & 0x01FFFFFFu) | (2u << 30);  // ga[56:32]|type=2
  i32x8 g1;
  g1[0] = 0x00020000;                        // data_size=2 (4 bytes)
  g1[1] = (int)((nd & 0xFFFFu) << 16);       // tensor_dim0[15:0]
  g1[2] = (int)(((nd >> 16) & 0xFFFFu) | (1u << 16));  // td0 hi | tensor_dim1=1
  g1[3] = (int)((nd & 0xFFFFu) << 16);       // tile_dim0
  g1[4] = 0;                                 // tile_dim1/2 unused
  g1[5] = (int)nd;                           // tensor_dim0_stride[31:0]
  g1[6] = 0;
  g1[7] = 0;
  i32x4 gz = {0, 0, 0, 0};
#if __clang_major__ >= 23
  i32x8 gz8 = {0, 0, 0, 0, 0, 0, 0, 0};
  __builtin_amdgcn_tensor_load_to_lds(g0, g1, gz, gz, gz8, 0);
#else
  __builtin_amdgcn_tensor_load_to_lds(g0, g1, gz, gz, 0);
#endif
  __builtin_amdgcn_s_wait_tensorcnt(0);
#else
  (void)src;
#endif
}

// ---------------------------------------------------------------------------
// Conv GEMM: one block per N-tile (16 out channels). 4 waves share the
// TDM-staged B operand in LDS and cover the 13 M-tiles. Inner loop:
// 2x global_load_b128 (A) + 2x ds_load_b128 (B) + v_wmma_f32_16x16x32_f16.
// ---------------------------------------------------------------------------
__global__ __launch_bounds__(128)
void frcnn_conv_gemm(const _Float16* __restrict__ packA,
                     const _Float16* __restrict__ packB,
                     const float* __restrict__ rpn_b,
                     float* __restrict__ grid) {
  __shared__ _Float16 sB[KC_CONV * 32 * 16];        // 73728 B
  const int ntile = blockIdx.x;
  const int tid   = threadIdx.x;
  const int wave  = tid >> 5;
  const int lane  = tid & 31;
  const int half  = lane >> 4, l15 = lane & 15;

#if HAVE_TDM
  if (wave == 0) tdm_load_to_lds0(packB + (size_t)ntile * (KC_CONV * 32 * 16));
  // The TDM descriptor carries the LDS address as an integer, so the compiler
  // cannot see that sB is written. Escape the pointer + memory clobber so the
  // subsequent sB loads are not folded to undef.
  asm volatile("" : : "v"((const void*)sB) : "memory");
#else
  {
    const uint4* src = (const uint4*)(packB + (size_t)ntile * (KC_CONV * 32 * 16));
    uint4* dst = (uint4*)sB;
    for (int i = tid; i < CONV_B_BYTES / 16; i += 128) dst[i] = src[i];
  }
#endif
  __syncthreads();

  const int oc = ntile * 16 + l15;
  const float bias = rpn_b[oc];
  for (int mt = wave; mt < MT_CONV; mt += 4) {
    const _Float16* aptr = packA + (size_t)(mt * KC_CONV * 32 + lane) * 16;
    v8f acc = {};
    for (int kc = 0; kc < KC_CONV; ++kc) {
      v16h a = *(const v16h*)(aptr + (size_t)kc * (32 * 16));
      v16h b = *(const v16h*)(sB + (size_t)(kc * 32 + lane) * 16);
      __builtin_prefetch(aptr + (size_t)(kc + 8) * (32 * 16), 0, 1);
      acc = __builtin_amdgcn_wmma_f32_16x16x32_f16(
          false, a, false, b, (short)0, acc, false, false);
    }
#pragma unroll
    for (int r = 0; r < 8; ++r)                 // unconditional: 208-padded grid
      grid[oc * GSTRIDE + mt * 16 + r + 8 * half] = acc[r] + bias;
  }
}

// ---------------------------------------------------------------------------
// 1x1 conv (36 ch) + sigmoid + anchor decode -> bboxes (4,1764)
// ---------------------------------------------------------------------------
__global__ __launch_bounds__(256)
void frcnn_offsets_bbox(const float* __restrict__ grid,
                        const float* __restrict__ off_w,
                        const float* __restrict__ off_b,
                        float* __restrict__ bboxes) {
  const int tid = blockIdx.x * blockDim.x + threadIdx.x;
  if (tid >= 36 * NPIX) return;
  const int oc = tid / NPIX, pix = tid % NPIX;
  float s = 0.f;
  for (int c = 0; c < HID; ++c) s += grid[c * GSTRIDE + pix] * off_w[oc * HID + c];
  s += off_b[oc];
  s = 1.0f / (1.0f + expf(-s));
  const int comp = oc / 9, a = oc % 9;
  const int y = pix / WW, x = pix % WW;
  const float scales[3] = {128.f / 600.f, 256.f / 600.f, 512.f / 600.f};
  const float sr[3]     = {0.70710678118f, 1.0f, 1.41421356237f};
  const int si = a / 3, ri = a % 3;
  float v;
  if (comp == 0)      v = (float)x / WW + s / WW;
  else if (comp == 1) v = (float)y / HH + s / HH;
  else if (comp == 2) v = s * scales[si] * sr[ri];
  else                v = s * scales[si] / sr[ri];
  bboxes[comp * NBOX + a * NPIX + pix] = v;
}

__global__ __launch_bounds__(256)
void frcnn_make_boxes(const float* __restrict__ bboxes, float* __restrict__ boxes) {
  const int n = blockIdx.x * blockDim.x + threadIdx.x;
  if (n >= NBOX) return;
  const float x1 = bboxes[0 * NBOX + n] * WW;
  const float y1 = bboxes[1 * NBOX + n] * HH;
  const float w  = bboxes[2 * NBOX + n] * WW;
  const float h  = bboxes[3 * NBOX + n] * HH;
  boxes[n * 4 + 0] = x1;
  boxes[n * 4 + 1] = y1;
  boxes[n * 4 + 2] = x1 + w;
  boxes[n * 4 + 3] = y1 + h;
}

// ---------------------------------------------------------------------------
// ROI max-pool (14x14 bins) + mean. Block per box, thread per channel.
// ---------------------------------------------------------------------------
__global__ __launch_bounds__(256)
void frcnn_roi_pool(const float* __restrict__ grid,
                    const float* __restrict__ boxes,
                    float* __restrict__ rois) {
  __shared__ int s_hs[14], s_he[14], s_ws[14], s_we[14];
  const int n = blockIdx.x;
  const int t = threadIdx.x;
  if (t < 14) {
    const int x1 = (int)rintf(boxes[n * 4 + 0]);   // RNE == jnp.round
    const int y1 = (int)rintf(boxes[n * 4 + 1]);
    const int x2 = (int)rintf(boxes[n * 4 + 2]);
    const int y2 = (int)rintf(boxes[n * 4 + 3]);
    const float bin_w = (float)max(x2 - x1 + 1, 1) / WW;
    const float bin_h = (float)max(y2 - y1 + 1, 1) / HH;
    const float pf = (float)t;
    s_hs[t] = min(max((int)floorf(pf * bin_h) + y1, 0), HH);
    s_he[t] = min(max((int)ceilf((pf + 1.f) * bin_h) + y1, 0), HH);
    s_ws[t] = min(max((int)floorf(pf * bin_w) + x1, 0), WW);
    s_we[t] = min(max((int)ceilf((pf + 1.f) * bin_w) + x1, 0), WW);
  }
  __syncthreads();
  const float* g = grid + t * GSTRIDE;
  float sum = 0.f;
  for (int ph = 0; ph < 14; ++ph) {
    float rowmax[14];
#pragma unroll
    for (int w = 0; w < 14; ++w) rowmax[w] = -INFINITY;
    for (int r = s_hs[ph]; r < s_he[ph]; ++r) {
#pragma unroll
      for (int w = 0; w < 14; ++w)
        rowmax[w] = fmaxf(rowmax[w], g[r * WW + w]);
    }
    for (int pw = 0; pw < 14; ++pw) {
      float m = -INFINITY;
      for (int w = s_ws[pw]; w < s_we[pw]; ++w) m = fmaxf(m, rowmax[w]);
      sum += (m == -INFINITY) ? 0.f : m;           // empty bin -> 0
    }
  }
  rois[n * HID + t] = sum * (1.0f / 196.0f);
}

// ---------------------------------------------------------------------------
// Head: hid = relu(rois @ h1_w^T + b1), scores = hid . h2_w + b2. 16 waves,
// one N-tile each, same 16-row M-tile; ds_add_f32 per-row reduction.
// ---------------------------------------------------------------------------
__global__ __launch_bounds__(512)
void frcnn_head_wmma(const _Float16* __restrict__ packA,
                     const _Float16* __restrict__ packB,
                     const float* __restrict__ h1_b,
                     const float* __restrict__ h2_w,
                     const float* __restrict__ h2_b,
                     float* __restrict__ scores) {
  __shared__ float accs[16];
  const int mtile = blockIdx.x;
  const int tid = threadIdx.x;
  if (tid < 16) accs[tid] = 0.f;
  __syncthreads();
  const int ntile = tid >> 5;
  const int lane  = tid & 31;
  const int half  = lane >> 4, l15 = lane & 15;
  v8f c = {};
#pragma unroll
  for (int kc = 0; kc < KC_HEAD; ++kc) {
    v16h a = *(const v16h*)(packA + (size_t)((mtile * KC_HEAD + kc) * 32 + lane) * 16);
    v16h b = *(const v16h*)(packB + (size_t)((ntile * KC_HEAD + kc) * 32 + lane) * 16);
    c = __builtin_amdgcn_wmma_f32_16x16x32_f16(
        false, a, false, b, (short)0, c, false, false);
  }
  const int col = ntile * 16 + l15;
  const float b1 = h1_b[col];
  const float w2 = h2_w[col];
#pragma unroll
  for (int r = 0; r < 8; ++r) {
    const float h = fmaxf(c[r] + b1, 0.f);
    atomicAdd(&accs[r + 8 * half], h * w2);        // ds_add_f32
  }
  __syncthreads();
  if (tid < 16) {
    const int orow = mtile * 16 + tid;
    if (orow < NBOX) scores[orow] = accs[tid] + h2_b[0];
  }
}

// ---------------------------------------------------------------------------
extern "C" void kernel_launch(void* const* d_in, const int* in_sizes, int n_in,
                              void* d_out, int out_size, void* d_ws, size_t ws_size,
                              hipStream_t stream) {
  const float* features = (const float*)d_in[0];
  const float* rpn_w    = (const float*)d_in[1];
  const float* rpn_b    = (const float*)d_in[2];
  const float* off_w    = (const float*)d_in[3];
  const float* off_b    = (const float*)d_in[4];
  const float* h1_w     = (const float*)d_in[5];
  const float* h1_b     = (const float*)d_in[6];
  const float* h2_w     = (const float*)d_in[7];
  const float* h2_b     = (const float*)d_in[8];
  float* out = (float*)d_out;            // [0,7056): bboxes ; [7056,8820): scores

  char* ws = (char*)d_ws;
  float*     grid   = (float*)ws;                       ws += (size_t)256 * GSTRIDE * 4;  // 212992 B
  float*     boxes  = (float*)ws;                       ws += (size_t)NBOX * 4 * 4;       //  28224 B
  float*     rois   = (float*)ws;                       ws += (size_t)NBOX * HID * 4;     // 1806336 B
  _Float16*  pA_cv  = (_Float16*)ws;                    ws += (size_t)MT_CONV * KC_CONV * 32 * 16 * 2;
  _Float16*  pB_cv  = (_Float16*)ws;                    ws += (size_t)16 * KC_CONV * 32 * 16 * 2;
  _Float16*  pB_h1  = (_Float16*)ws;                    ws += (size_t)16 * KC_HEAD * 32 * 16 * 2;
  _Float16*  pA_hd  = (_Float16*)ws;                    // MT_HEAD*KC_HEAD*32*16*2 B

  pack_convA<<<(MT_CONV * KC_CONV * 32 + 255) / 256, 256, 0, stream>>>(features, pA_cv);
  pack_convB<<<(16 * KC_CONV * 32 + 255) / 256, 256, 0, stream>>>(rpn_w, pB_cv);
  pack_h1B  <<<(16 * KC_HEAD * 32 + 255) / 256, 256, 0, stream>>>(h1_w, pB_h1);

  frcnn_conv_gemm<<<16, 128, 0, stream>>>(pA_cv, pB_cv, rpn_b, grid);
  frcnn_offsets_bbox<<<(36 * NPIX + 255) / 256, 256, 0, stream>>>(grid, off_w, off_b, out);
  frcnn_make_boxes<<<(NBOX + 255) / 256, 256, 0, stream>>>(out, boxes);
  frcnn_roi_pool<<<NBOX, 256, 0, stream>>>(grid, boxes, rois);
  pack_headA<<<(MT_HEAD * KC_HEAD * 32 + 255) / 256, 256, 0, stream>>>(rois, pA_hd);
  frcnn_head_wmma<<<MT_HEAD, 512, 0, stream>>>(pA_hd, pB_h1, h1_b, h2_w, h2_b, out + 4 * NBOX);
}
